// LIFBenchmark_944892805201
// MI455X (gfx1250) — matrix operand order repaired and checked by reading, not compiled, not run on hardware
//
#include <hip/hip_runtime.h>

typedef __attribute__((ext_vector_type(16))) __bf16 v16bf;
typedef __attribute__((ext_vector_type(8)))  float  v8f;

#define T_STEPS 1024
#define BATCH   128
#define FDIM    512
#define LDS_STRIDE 520   // bf16 per row, padded vs 512 to dodge bank conflicts

__global__ __launch_bounds__(128)
void lif_fused_wmma_kernel(const float* __restrict__ S,
                           const float* __restrict__ W,
                           float* __restrict__ out)
{
    // double-buffered spike tile: [buf][16 batch rows][512 f cols] bf16
    __shared__ __align__(16) __bf16 As[2][16][LDS_STRIDE];

    const int tid  = threadIdx.x;
    const int lane = tid & 31;
    const int wave = tid >> 5;            // 0..3 (wave32)
    const int lo   = lane & 15;
    const int hi   = lane >> 4;           // 0 or 1

    const int btile = blockIdx.x >> 3;        // 0..7 : 16-row batch tile
    const int go    = blockIdx.x & 7;         // 0..7 : 64-wide g octant
    const int gbase = go * 64 + wave * 16;    // this wave's 16 output columns

    // ---- preload B fragments (W^T slice, 512x16) as bf16 into registers ----
    // B frag k, lane(lo,hi): col g = gbase+lo, K = 32k + 16*hi .. +16 (contiguous)
    v16bf wfrag[16];
    {
        const int g = gbase + lo;
        const float* wrow = W + (size_t)g * FDIM;
        #pragma unroll
        for (int k = 0; k < 16; ++k) {
            const float* p = wrow + k * 32 + hi * 16;
            #pragma unroll
            for (int e = 0; e < 16; ++e)
                wfrag[k][e] = (__bf16)p[e];
        }
    }

    // LIF state in f32 accumulator layout: element r = (m = r+8*hi, n = lo)
    v8f v = {}, cur = {}, z = {};

    const size_t tile_base = (size_t)btile * 16 * FDIM;  // contiguous 16x512 slab

    // ---- prologue: stage S[0] into buffer 0 ----
    {
        const float* slab = S + tile_base;
        #pragma unroll
        for (int r = 0; r < 16; ++r) {
            float4 q = reinterpret_cast<const float4*>(slab + (size_t)r * FDIM)[tid];
            __bf16* dst = &As[0][r][tid * 4];
            dst[0] = (__bf16)q.x; dst[1] = (__bf16)q.y;
            dst[2] = (__bf16)q.z; dst[3] = (__bf16)q.w;
        }
    }
    __syncthreads();

    for (int t = 0; t < T_STEPS; ++t) {
        const int cb = t & 1;        // buffer to compute from
        const int nb = cb ^ 1;       // buffer being filled for t+1
        const bool have_next = (t + 1 < T_STEPS);

        // ---- issue global loads for S[t+1] early (latency hidden by WMMAs) ----
        float4 q[16];
        if (have_next) {
            const float* slab = S + (size_t)(t + 1) * (BATCH * FDIM) + tile_base;
            #pragma unroll
            for (int r = 0; r < 16; ++r)
                q[r] = reinterpret_cast<const float4*>(slab + (size_t)r * FDIM)[tid];
            if (t + 2 < T_STEPS)   // warm L2 for t+2 (global_prefetch_b8)
                __builtin_prefetch((const char*)(S + (size_t)(t + 2) * (BATCH * FDIM) + tile_base)
                                       + tid * 256, 0, 1);
        }

        // ---- x_t tile = A(16x512 bf16) x B(512x16 bf16), f32 accumulate ----
        v8f c0 = {}, c1 = {};
        #pragma unroll
        for (int k = 0; k < 16; k += 2) {
            // A frag k, lane(lo,hi): row lo, K = 32k + 16*hi .. +16 (contiguous)
            union { uint4 u[2]; v16bf bv; } a0, a1;
            const __bf16* r0 = &As[cb][lo][k * 32 + hi * 16];
            a0.u[0] = *reinterpret_cast<const uint4*>(r0);
            a0.u[1] = *reinterpret_cast<const uint4*>(r0 + 8);
            const __bf16* r1 = &As[cb][lo][(k + 1) * 32 + hi * 16];
            a1.u[0] = *reinterpret_cast<const uint4*>(r1);
            a1.u[1] = *reinterpret_cast<const uint4*>(r1 + 8);
            c0 = __builtin_amdgcn_wmma_f32_16x16x32_bf16(false, a0.bv, false, wfrag[k],
                                                         (short)0, c0, false, false);
            c1 = __builtin_amdgcn_wmma_f32_16x16x32_bf16(false, a1.bv, false, wfrag[k + 1],
                                                         (short)0, c1, false, false);
        }
        v8f x = c0 + c1;

        // ---- drain staged loads: fp32 -> bf16 into the other buffer ----
        if (have_next) {
            #pragma unroll
            for (int r = 0; r < 16; ++r) {
                __bf16* dst = &As[nb][r][tid * 4];
                dst[0] = (__bf16)q[r].x; dst[1] = (__bf16)q[r].y;
                dst[2] = (__bf16)q[r].z; dst[3] = (__bf16)q[r].w;
            }
        }

        // ---- LIF update (V_LEAK=0, V_RESET=0) ----
        #pragma unroll
        for (int r = 0; r < 8; ++r) {
            float vd = v[r] + 0.1f * (cur[r] - v[r]);   // DT*TAU_MEM_INV = 0.1
            float id = 0.8f * cur[r];                   // 1 - DT*TAU_SYN_INV = 0.8
            float zz = (vd - 1.0f) > 0.0f ? 1.0f : 0.0f;
            z[r]   = zz;
            v[r]   = (1.0f - zz) * vd;
            cur[r] = id + x[r];
        }

        // single barrier per step: reads of As[cb] and writes of As[nb] both
        // complete before any wave proceeds to t+1
        __syncthreads();
    }

    // ---- outputs: z, z, v, i each [128,512] fp32, concatenated ----
    const int g = gbase + lo;
    #pragma unroll
    for (int r = 0; r < 8; ++r) {
        int    b   = btile * 16 + r + hi * 8;
        size_t idx = (size_t)b * FDIM + g;
        out[idx]          = z[r];
        out[ 65536 + idx] = z[r];
        out[131072 + idx] = v[r];
        out[196608 + idx] = cur[r];
    }
}

extern "C" void kernel_launch(void* const* d_in, const int* in_sizes, int n_in,
                              void* d_out, int out_size, void* d_ws, size_t ws_size,
                              hipStream_t stream) {
    (void)in_sizes; (void)n_in; (void)d_ws; (void)ws_size; (void)out_size;
    const float* S = (const float*)d_in[0];   // [1024,128,512] fp32
    const float* W = (const float*)d_in[1];   // [512,512] fp32
    float* out = (float*)d_out;               // z,z,v,i -> 4*128*512 fp32

    dim3 grid(64);    // 8 batch tiles x 8 g-octants
    dim3 block(128);  // 4 wave32 per workgroup
    lif_fused_wmma_kernel<<<grid, block, 0, stream>>>(S, W, out);
}
